// DeformableConv2d_63471026700757
// MI455X (gfx1250) — matrix operand order repaired and checked by reading, not compiled, not run on hardware
//
#include <hip/hip_runtime.h>
#include <hip/hip_bf16.h>
#include <math.h>

typedef __attribute__((ext_vector_type(16))) _Float16 v16h;
typedef __attribute__((ext_vector_type(8)))  _Float16 v8h;
typedef __attribute__((ext_vector_type(8)))  float    v8f;
typedef __attribute__((ext_vector_type(4)))  float    v4f;

#define BB 8
#define CC 128
#define OO 256
#define HH 96
#define WWID 96
#define HW (HH * WWID)
#define K9 9
#define CK (CC * K9)   // 1152
#define LDA 40         // LDS row stride (halfs), 80B = 16B aligned, conflict-padded
#define LDB 40

static __device__ __forceinline__ int iclamp(int v, int lo, int hi) {
    return v < lo ? lo : (v > hi ? hi : v);
}

// ---------------------------------------------------------------------------
// Kernel 1: NCHW -> NHWC transpose of x (tiled via LDS, coalesced both ways)
// grid (HW/32, C/32, B), block 256
// ---------------------------------------------------------------------------
__global__ __launch_bounds__(256) void nchw_to_nhwc(const float* __restrict__ x,
                                                    float* __restrict__ xn) {
    __shared__ float tile[32][33];
    const int tx = threadIdx.x & 31;   // p on load, c on store
    const int ty = threadIdx.x >> 5;   // 0..7
    const int b = blockIdx.z;
    const int pBase = blockIdx.x * 32;
    const int cBase = blockIdx.y * 32;
#pragma unroll
    for (int i = 0; i < 4; ++i) {
        const int c = cBase + ty + i * 8;
        tile[ty + i * 8][tx] = x[(size_t)(b * CC + c) * HW + pBase + tx];
    }
    __syncthreads();
#pragma unroll
    for (int i = 0; i < 4; ++i) {
        const int p = pBase + ty + i * 8;
        xn[(size_t)(b * HW + p) * CC + cBase + tx] = tile[tx][ty + i * 8];
    }
}

// ---------------------------------------------------------------------------
// Kernel 2a: pack offset_w (18ch) + mask_w (9ch) into wcat[tap][c][27]
// ---------------------------------------------------------------------------
__global__ void pack_wcat(const float* __restrict__ ow, const float* __restrict__ mw,
                          float* __restrict__ wcat) {
    const int i = blockIdx.x * 256 + threadIdx.x;   // over CK*27
    if (i >= CK * 27) return;
    const int co = i % 27;
    const int tc = i / 27;          // tap*128 + c
    const int tap = tc / CC;
    const int c = tc % CC;
    float v;
    if (co < 18) v = ow[((size_t)co * CC + c) * K9 + tap];
    else         v = mw[((size_t)(co - 18) * CC + c) * K9 + tap];
    wcat[i] = v;
}

// ---------------------------------------------------------------------------
// Kernel 2b: pack main weight [O,C,9] fp32 -> f16 [O][tap*128+c] (GEMM A)
// ---------------------------------------------------------------------------
__global__ void pack_wf16(const float* __restrict__ wt, _Float16* __restrict__ wf) {
    const int i = blockIdx.x * 256 + threadIdx.x;   // over O*CK
    if (i >= OO * CK) return;
    const int o = i / CK;
    const int k = i % CK;
    const int tap = k / CC;
    const int c = k % CC;
    wf[i] = (_Float16)wt[((size_t)o * CC + c) * K9 + tap];
}

// ---------------------------------------------------------------------------
// Kernel 3: fused offset+mask 3x3 conv -> absolute coords py/px + sigmoid mask
// one wave per output pixel, lanes split the 128 input channels
// grid (B*HW/8), block 256
// ---------------------------------------------------------------------------
__global__ __launch_bounds__(256) void offmask_conv(const float* __restrict__ xn,
                                                    const float* __restrict__ wcat,
                                                    const float* __restrict__ ob,
                                                    const float* __restrict__ mb,
                                                    float* __restrict__ py,
                                                    float* __restrict__ px,
                                                    float* __restrict__ mk) {
    const int lane = threadIdx.x & 31;
    const int wave = threadIdx.x >> 5;
    const int pix = blockIdx.x * 8 + wave;          // 0 .. B*HW-1
    const int b = pix / HW;
    const int p = pix % HW;
    const int h = p / WWID, w = p % WWID;

    float acc[27];
#pragma unroll
    for (int i = 0; i < 27; ++i) acc[i] = 0.0f;

    for (int tap = 0; tap < 9; ++tap) {
        const int ny = h - 1 + tap / 3;
        const int nx = w - 1 + tap % 3;
        const bool inb = (ny >= 0) & (ny < HH) & (nx >= 0) & (nx < WWID);
        const float* xp = xn + ((size_t)b * HW +
                                iclamp(ny, 0, HH - 1) * WWID + iclamp(nx, 0, WWID - 1)) * CC;
#pragma unroll
        for (int c4 = 0; c4 < 4; ++c4) {
            const int c = lane + c4 * 32;
            const float xv = inb ? xp[c] : 0.0f;
            const float* wp = wcat + (size_t)(tap * CC + c) * 27;
#pragma unroll
            for (int co = 0; co < 27; ++co)
                acc[co] = fmaf(xv, wp[co], acc[co]);
        }
    }
    // butterfly reduce each of the 27 sums across the wave32
#pragma unroll
    for (int co = 0; co < 27; ++co)
#pragma unroll
        for (int s = 16; s > 0; s >>= 1)
            acc[co] += __shfl_xor(acc[co], s, 32);

    if (lane < 18) {
        const int tap = lane >> 1;
        const float v = acc[lane] + ob[lane];
        const size_t idx = (size_t)(b * K9 + tap) * HW + p;
        if ((lane & 1) == 0) py[idx] = (float)(h - 1 + tap / 3) + v;   // 2k   -> y
        else                 px[idx] = (float)(w - 1 + tap % 3) + v;   // 2k+1 -> x
    } else if (lane < 27) {
        const int tap = lane - 18;
        const float v = acc[lane] + mb[tap];
        mk[(size_t)(b * K9 + tap) * HW + p] = 1.0f / (1.0f + expf(-v));
    }
}

// ---------------------------------------------------------------------------
// Kernel 4: fused deformable-im2col + WMMA GEMM (full-O tile per workgroup)
//   out[b, 0..256, nBase+64] = w_f16[256 x CK] * col_f16[CK x 64]
// block = 256 threads = 8 waves; wave wv owns M rows [wv*32, wv*32+32), all 64 N
// tap-outer / c-chunk-inner K loop: sampling coords computed once per tap
// grid (HW/64, 1, B)
// ---------------------------------------------------------------------------
__global__ __launch_bounds__(256) void dcn_wmma_gemm(const float* __restrict__ xn,
                                                     const float* __restrict__ py,
                                                     const float* __restrict__ px,
                                                     const float* __restrict__ mk,
                                                     const _Float16* __restrict__ wf,
                                                     const float* __restrict__ bias,
                                                     float* __restrict__ out) {
    __shared__ _Float16 As[256 * LDA];   // weights tile  [256 M][32 K]
    __shared__ _Float16 Bs[64 * LDB];    // im2col tile   [64 N][32 K] (N-major)

    const int t = threadIdx.x;
    const int b = blockIdx.z;
    const int nBase = blockIdx.x * 64;

    const int bpix = t >> 2;             // B build: pixel 0..63
    const int bsub = (t & 3) * 8;        // B build: channel sub-offset 0/8/16/24

    const int lane = t & 31;
    const int wv = t >> 5;
    const int wm = wv * 32;              // wave M offset (8 waves x 32 rows = 256)

    const int p = nBase + bpix;          // flattened output pixel for B build

    v8f acc[2][4] = {};                  // [mi][ni] : 32(M) x 64(N) per wave

    for (int tap = 0; tap < 9; ++tap) {
        // ---- per-tap sampling state (reused across the 4 channel chunks) ----
        const size_t sidx = (size_t)(b * K9 + tap) * HW + p;
        const float sy = py[sidx];
        const float sx = px[sidx];
        const float sm = mk[sidx];
        const float y0f = floorf(sy);
        const float x0f = floorf(sx);
        const int y0 = (int)y0f, x0 = (int)x0f;
        const int y1 = y0 + 1, x1 = x0 + 1;
        const float wy1 = sy - y0f, wy0 = 1.0f - wy1;
        const float wx1 = sx - x0f, wx0 = 1.0f - wx1;
        const bool vy0 = (y0 >= 0) & (y0 < HH);
        const bool vy1 = (y1 >= 0) & (y1 < HH);
        const bool vx0 = (x0 >= 0) & (x0 < WWID);
        const bool vx1 = (x1 >= 0) & (x1 < WWID);
        const float w00 = (vy0 & vx0) ? wy0 * wx0 * sm : 0.0f;
        const float w01 = (vy0 & vx1) ? wy0 * wx1 * sm : 0.0f;
        const float w10 = (vy1 & vx0) ? wy1 * wx0 * sm : 0.0f;
        const float w11 = (vy1 & vx1) ? wy1 * wx1 * sm : 0.0f;

        const float* cb = xn + (size_t)b * HW * CC + bsub;
        const float* p00 = cb + (size_t)(iclamp(y0, 0, HH - 1) * WWID + iclamp(x0, 0, WWID - 1)) * CC;
        const float* p01 = cb + (size_t)(iclamp(y0, 0, HH - 1) * WWID + iclamp(x1, 0, WWID - 1)) * CC;
        const float* p10 = cb + (size_t)(iclamp(y1, 0, HH - 1) * WWID + iclamp(x0, 0, WWID - 1)) * CC;
        const float* p11 = cb + (size_t)(iclamp(y1, 0, HH - 1) * WWID + iclamp(x1, 0, WWID - 1)) * CC;

#pragma unroll 1
        for (int c4 = 0; c4 < 4; ++c4) {
            const int cBase = c4 * 32;

            // ---- stage A tile: one weight row per thread, 32 halfs ----
            {
                const _Float16* src = wf + (size_t)t * CK + tap * CC + cBase;
                const v8h w0 = *(const v8h*)(src);
                const v8h w1 = *(const v8h*)(src + 8);
                const v8h w2 = *(const v8h*)(src + 16);
                const v8h w3 = *(const v8h*)(src + 24);
                *(v8h*)(&As[t * LDA])      = w0;
                *(v8h*)(&As[t * LDA + 8])  = w1;
                *(v8h*)(&As[t * LDA + 16]) = w2;
                *(v8h*)(&As[t * LDA + 24]) = w3;
                __builtin_prefetch(src + 32, 0, 3);
            }

            // ---- build B tile: modulated bilinear samples -> f16 LDS ----
            {
                const v4f a00 = *(const v4f*)(p00 + cBase), b00 = *(const v4f*)(p00 + cBase + 4);
                const v4f a01 = *(const v4f*)(p01 + cBase), b01 = *(const v4f*)(p01 + cBase + 4);
                const v4f a10 = *(const v4f*)(p10 + cBase), b10 = *(const v4f*)(p10 + cBase + 4);
                const v4f a11 = *(const v4f*)(p11 + cBase), b11 = *(const v4f*)(p11 + cBase + 4);

                _Float16* dst = &Bs[bpix * LDB + bsub];
#pragma unroll
                for (int j = 0; j < 4; ++j) {
                    dst[j]     = (_Float16)(w00 * a00[j] + w01 * a01[j] + w10 * a10[j] + w11 * a11[j]);
                    dst[j + 4] = (_Float16)(w00 * b00[j] + w01 * b01[j] + w10 * b10[j] + w11 * b11[j]);
                }
            }
            __syncthreads();

            // ---- WMMA: fragments per the CDNA5 16-bit A/B VGPR layouts ----
            {
                const int klo = (lane >> 4) * 8;   // A: lanes0-15 K0..7/16..23, lanes16-31 K8..15/24..31
                const int kbo = (lane >> 4) * 8;   // B: lanes0-15 K0..7/16..23, lanes16-31 K8..15/24..31
                const int row = lane & 15;

                v16h bfrag[4];
#pragma unroll
                for (int ni = 0; ni < 4; ++ni) {
                    const _Float16* bp = &Bs[(ni * 16 + row) * LDB + kbo];
                    const v8h blo = *(const v8h*)(bp);
                    const v8h bhi = *(const v8h*)(bp + 16);
                    bfrag[ni] = __builtin_shufflevector(blo, bhi, 0, 1, 2, 3, 4, 5, 6, 7,
                                                        8, 9, 10, 11, 12, 13, 14, 15);
                }
#pragma unroll
                for (int mi = 0; mi < 2; ++mi) {
                    const _Float16* ap = &As[(wm + mi * 16 + row) * LDA];
                    const v8h alo = *(const v8h*)(ap + klo);
                    const v8h ahi = *(const v8h*)(ap + 16 + klo);
                    const v16h afrag = __builtin_shufflevector(alo, ahi, 0, 1, 2, 3, 4, 5, 6, 7,
                                                               8, 9, 10, 11, 12, 13, 14, 15);
#pragma unroll
                    for (int ni = 0; ni < 4; ++ni)
                        acc[mi][ni] = __builtin_amdgcn_wmma_f32_16x16x32_f16(
                            false, afrag, false, bfrag[ni],
                            (short)0, acc[mi][ni], false, false);
                }
            }
            __syncthreads();
        }
    }

    // ---- epilogue: C layout (VGPR r -> M=r / r+8 by lane half; N = lane&15) ----
#pragma unroll
    for (int mi = 0; mi < 2; ++mi)
#pragma unroll
        for (int ni = 0; ni < 4; ++ni) {
            const int o0 = wm + mi * 16 + ((lane >> 4) * 8);
            const int n = nBase + ni * 16 + (lane & 15);
#pragma unroll
            for (int r = 0; r < 8; ++r) {
                const int o = o0 + r;
                out[(size_t)(b * OO + o) * HW + n] = acc[mi][ni][r] + bias[o];
            }
        }
}

// ---------------------------------------------------------------------------
extern "C" void kernel_launch(void* const* d_in, const int* in_sizes, int n_in,
                              void* d_out, int out_size, void* d_ws, size_t ws_size,
                              hipStream_t stream) {
    const float* x  = (const float*)d_in[0];
    const float* ow = (const float*)d_in[1];
    const float* ob = (const float*)d_in[2];
    const float* mw = (const float*)d_in[3];
    const float* mb = (const float*)d_in[4];
    const float* wt = (const float*)d_in[5];
    const float* bias = (const float*)d_in[6];
    float* out = (float*)d_out;

    char* ws = (char*)d_ws;
    size_t off = 0;
    float* xn = (float*)(ws + off);   off += (size_t)BB * HW * CC * sizeof(float);   // 37.7 MB
    float* py = (float*)(ws + off);   off += (size_t)BB * K9 * HW * sizeof(float);   // 2.65 MB
    float* px = (float*)(ws + off);   off += (size_t)BB * K9 * HW * sizeof(float);
    float* mk = (float*)(ws + off);   off += (size_t)BB * K9 * HW * sizeof(float);
    float* wcat = (float*)(ws + off); off += (size_t)CK * 27 * sizeof(float);        // 124 KB
    _Float16* wf = (_Float16*)(ws + off);                                            // 576 KB

    // 1) NCHW -> NHWC
    nchw_to_nhwc<<<dim3(HW / 32, CC / 32, BB), 256, 0, stream>>>(x, xn);
    // 2) weight repacks
    pack_wcat<<<(CK * 27 + 255) / 256, 256, 0, stream>>>(ow, mw, wcat);
    pack_wf16<<<(OO * CK + 255) / 256, 256, 0, stream>>>(wt, wf);
    // 3) offset/mask conv -> absolute sample coords + sigmoid mask
    offmask_conv<<<(BB * HW) / 8, 256, 0, stream>>>(xn, wcat, ob, mb, py, px, mk);
    // 4) fused deformable-im2col + WMMA GEMM (full O per workgroup)
    dcn_wmma_gemm<<<dim3(HW / 64, 1, BB), 256, 0, stream>>>(xn, py, px, mk, wf, bias, out);
}
